// RoutingTransfomer_71536975282843
// MI455X (gfx1250) — matrix operand order repaired and checked by reading, not compiled, not run on hardware
//
#include <hip/hip_runtime.h>

// ---------------- dims ----------------
constexpr int Bb = 8, SP = 1024, SF = 1024, XD = 6;
constexpr int Dd = 512, Ll = 4, Hh = 8, TW = 128;     // TW = tokens per cluster
constexpr int DH = Dd / Hh;                            // 64
constexpr int Ss = SP + SF;                            // 2048
constexpr int CL = Ss / TW;                            // 16 clusters
constexpr int FF = 4 * Dd;                             // 2048
constexpr int BS = Bb * Ss;                            // 16384 rows

typedef __bf16 bf16_t;
typedef __attribute__((ext_vector_type(16))) __bf16 v16bf;
typedef __attribute__((ext_vector_type(8)))  __bf16 v8bf;
typedef __attribute__((ext_vector_type(8)))  float  v8f;

static __device__ __forceinline__ bf16_t f2bf(float f) {
    union { float f; unsigned u; } v; v.f = f;
    unsigned r = (v.u + 0x7FFFu + ((v.u >> 16) & 1u)) >> 16;
    unsigned short s = (unsigned short)r;
    bf16_t o; __builtin_memcpy(&o, &s, 2);
    return o;
}

// ---------------- embedding: h = concat(x)@W_emb + b_emb ----------------
__global__ void embed_kernel(const float* __restrict__ past_x, const float* __restrict__ past_y,
                             const float* __restrict__ future_x, const float* __restrict__ W_emb,
                             const float* __restrict__ b_emb, float* __restrict__ h) {
    long idx = (long)blockIdx.x * blockDim.x + threadIdx.x;
    if (idx >= (long)BS * Dd) return;
    int dcol = (int)(idx % Dd);
    long row = idx / Dd;
    int s = (int)(row % Ss);
    int b = (int)(row / Ss);
    float xv[7];
    if (s < SP) {
        const float* px = past_x + ((long)(b * SP + s)) * XD;
        #pragma unroll
        for (int j = 0; j < XD; ++j) xv[j] = px[j];
        xv[XD] = past_y[b * SP + s];
    } else {
        int sf = s - SP;
        const float* fx = future_x + ((long)(b * SF + sf)) * XD;
        #pragma unroll
        for (int j = 0; j < XD; ++j) xv[j] = fx[j];
        xv[XD] = past_y[b * SP + (SP - 1)];   // repeated last observed y
    }
    float acc = b_emb[dcol];
    #pragma unroll
    for (int j = 0; j < XD + 1; ++j) acc += xv[j] * W_emb[j * Dd + dcol];
    h[idx] = acc;
}

// ---------------- layernorm over D=512 (one wave32 per row) ----------------
__global__ void ln_kernel(const float* __restrict__ x, const float* __restrict__ g,
                          const float* __restrict__ bvec, float* __restrict__ outF,
                          bf16_t* __restrict__ out16, int to16) {
    int wave = threadIdx.x >> 5, lane = threadIdx.x & 31;
    long row = (long)blockIdx.x * 8 + wave;
    const float* xr = x + row * Dd;
    float vals[16]; float s = 0.f, sq = 0.f;
    #pragma unroll
    for (int j = 0; j < 16; ++j) { float v = xr[lane + 32 * j]; vals[j] = v; s += v; sq += v * v; }
    #pragma unroll
    for (int o = 16; o > 0; o >>= 1) { s += __shfl_xor(s, o, 32); sq += __shfl_xor(sq, o, 32); }
    float mean = s * (1.0f / Dd);
    float var  = sq * (1.0f / Dd) - mean * mean;
    float r = rsqrtf(var + 1e-5f);
    #pragma unroll
    for (int j = 0; j < 16; ++j) {
        int i = lane + 32 * j;
        float y = (vals[j] - mean) * r * g[i] + bvec[i];
        if (to16) out16[row * Dd + i] = f2bf(y);
        else      outF[row * Dd + i] = y;
    }
}

// ---------------- fp32 (K x N) -> bf16 transposed (N x K) ----------------
__global__ void transpose_bf16_kernel(const float* __restrict__ in, bf16_t* __restrict__ out,
                                      int K, int N) {
    long idx = (long)blockIdx.x * blockDim.x + threadIdx.x;
    if (idx >= (long)K * N) return;
    int k = (int)(idx % K);
    int n = (int)(idx / K);
    out[idx] = f2bf(in[(long)k * N + n]);
}

// -------- WMMA GEMM, 32x32 tile per wave (2x2 fragments, 4 accumulators) ----
// C[M,N] = A[M,K] @ Bt[N,K]^T (+epilogue)
// flags: 1=bias, 2=gelu(tanh approx), 4=residual add into Cf, 8=store bf16 C16
__global__ void gemm_wmma_kernel(const bf16_t* __restrict__ A, const bf16_t* __restrict__ Bt,
                                 float* __restrict__ Cf, bf16_t* __restrict__ C16,
                                 const float* __restrict__ bias,
                                 int M, int N, int K,
                                 long sA, long sB, long sC, int flags) {
    int bz = blockIdx.z;
    A  += (long)bz * sA;
    Bt += (long)bz * sB;
    long cOff = (long)bz * sC;
    int wave = threadIdx.x >> 5, lane = threadIdx.x & 31;
    int tm = wave & 1, tn = wave >> 1;                 // 8 waves = 2(M) x 4(N), 32x32 each
    int row0 = blockIdx.y * 64 + tm * 32;
    int col0 = blockIdx.x * 128 + tn * 32;
    if (row0 >= M || col0 >= N) return;
    int half = lane >> 4;                              // lane-half selects K sub-range
    int mr   = lane & 15;                              // M (for A) / N (for B) index
    const bf16_t* arow0 = A  + (long)(row0 + mr) * K + half * 8;
    const bf16_t* arow1 = arow0 + (long)16 * K;
    const bf16_t* brow0 = Bt + (long)(col0 + mr) * K + half * 8;
    const bf16_t* brow1 = brow0 + (long)16 * K;
    v8f acc[2][2] = {};
    for (int k0 = 0; k0 < K; k0 += 32) {
        __builtin_prefetch(arow0 + k0 + 128, 0, 3);
        __builtin_prefetch(arow1 + k0 + 128, 0, 3);
        __builtin_prefetch(brow0 + k0 + 128, 0, 3);
        __builtin_prefetch(brow1 + k0 + 128, 0, 3);
        v8bf a0lo = *(const v8bf*)(arow0 + k0);        // K = k0 + half*8 + [0..7]
        v8bf a0hi = *(const v8bf*)(arow0 + k0 + 16);   // K = k0 + 16 + half*8 + [0..7]
        v8bf a1lo = *(const v8bf*)(arow1 + k0);
        v8bf a1hi = *(const v8bf*)(arow1 + k0 + 16);
        v16bf a0, a1;
        #pragma unroll
        for (int i = 0; i < 8; ++i) {
            a0[i] = a0lo[i]; a0[8 + i] = a0hi[i];
            a1[i] = a1lo[i]; a1[8 + i] = a1hi[i];
        }
        v8bf b0lo = *(const v8bf*)(brow0 + k0);        // K = k0 + half*8 + [0..7]
        v8bf b0hi = *(const v8bf*)(brow0 + k0 + 16);   // K = k0 + 16 + half*8 + [0..7]
        v8bf b1lo = *(const v8bf*)(brow1 + k0);
        v8bf b1hi = *(const v8bf*)(brow1 + k0 + 16);
        v16bf b0, b1;
        #pragma unroll
        for (int i = 0; i < 8; ++i) {
            b0[i] = b0lo[i]; b0[8 + i] = b0hi[i];
            b1[i] = b1lo[i]; b1[8 + i] = b1hi[i];
        }
        acc[0][0] = __builtin_amdgcn_wmma_f32_16x16x32_bf16(false, a0, false, b0,
                                                            (short)0, acc[0][0], false, false);
        acc[0][1] = __builtin_amdgcn_wmma_f32_16x16x32_bf16(false, a0, false, b1,
                                                            (short)0, acc[0][1], false, false);
        acc[1][0] = __builtin_amdgcn_wmma_f32_16x16x32_bf16(false, a1, false, b0,
                                                            (short)0, acc[1][0], false, false);
        acc[1][1] = __builtin_amdgcn_wmma_f32_16x16x32_bf16(false, a1, false, b1,
                                                            (short)0, acc[1][1], false, false);
    }
    #pragma unroll
    for (int ti = 0; ti < 2; ++ti) {
        #pragma unroll
        for (int tj = 0; tj < 2; ++tj) {
            int n = col0 + tj * 16 + mr;
            #pragma unroll
            for (int r = 0; r < 8; ++r) {
                int m = row0 + ti * 16 + r + half * 8;
                float v = acc[ti][tj][r];
                if (flags & 1) v += bias[n];
                if (flags & 2) {
                    float x = v;
                    float t = tanhf(0.7978845608f * (x + 0.044715f * x * x * x));
                    v = 0.5f * x * (1.0f + t);
                }
                long idx = cOff + (long)m * N + n;
                if (flags & 4) v += Cf[idx];
                if (flags & 8) C16[idx] = f2bf(v);
                else           Cf[idx] = v;
            }
        }
    }
}

// ---------------- routing scores: scores[b,h,c,s] = normalize(q)·means[h,c] --
__global__ void route_scores_kernel(const float* __restrict__ qf, const float* __restrict__ means,
                                    float* __restrict__ scores) {
    long idx = (long)blockIdx.x * blockDim.x + threadIdx.x;
    if (idx >= (long)Bb * Hh * Ss) return;
    int s  = (int)(idx % Ss);
    int bh = (int)(idx / Ss);
    int hh = bh % Hh, b = bh / Hh;
    const float* qr = qf + ((long)(b * Ss + s)) * Dd + hh * DH;
    float q[DH]; float ssum = 1e-8f;
    #pragma unroll
    for (int d = 0; d < DH; ++d) { float v = qr[d]; q[d] = v; ssum += v * v; }
    float inv = rsqrtf(ssum);
    for (int c = 0; c < CL; ++c) {
        const float* mm = means + ((long)(hh * CL + c)) * DH;
        float dot = 0.f;
        #pragma unroll
        for (int d = 0; d < DH; ++d) dot += q[d] * mm[d];
        scores[(((long)(b * Hh + hh) * CL + c) * Ss) + s] = dot * inv;
    }
}

// ---------------- exact top-W per (b,h,c) by rank counting ----------------
__global__ void topk_kernel(const float* __restrict__ scores, int* __restrict__ idxout) {
    __shared__ float sc[Ss];
    long base = (long)blockIdx.x * Ss;
    for (int i = threadIdx.x; i < Ss; i += 256) sc[i] = scores[base + i];
    __syncthreads();
    for (int i = threadIdx.x; i < Ss; i += 256) {
        float v = sc[i];
        int rank = 0;
        for (int j = 0; j < Ss; ++j) {
            float u = sc[j];
            rank += (u > v) || (u == v && j < i);      // tiebreak: lower index first
        }
        if (rank < TW) idxout[(long)blockIdx.x * TW + rank] = i;
    }
}

// ---------------- gather q/k/v into per-cluster bf16 tiles ----------------
__global__ void gather_kernel(const float* __restrict__ qf, const float* __restrict__ kf,
                              const float* __restrict__ vf, const int* __restrict__ iq,
                              const int* __restrict__ ik, bf16_t* __restrict__ qg,
                              bf16_t* __restrict__ kg, bf16_t* __restrict__ vgT) {
    long idx = (long)blockIdx.x * blockDim.x + threadIdx.x;
    if (idx >= (long)Bb * Hh * CL * TW * DH) return;
    int d = (int)(idx & (DH - 1));
    long t = idx >> 6;
    int w = (int)(t % TW); t /= TW;
    int c = (int)(t % CL); t /= CL;
    int hh = (int)(t % Hh);
    int b  = (int)(t / Hh);
    long bhc = ((long)(b * Hh + hh) * CL + c);
    int tq = iq[bhc * TW + w];
    int tk = ik[bhc * TW + w];
    long qoff = ((long)(b * Ss + tq)) * Dd + hh * DH + d;
    long koff = ((long)(b * Ss + tk)) * Dd + hh * DH + d;
    qg[(bhc * TW + w) * DH + d] = f2bf(qf[qoff]);
    kg[(bhc * TW + w) * DH + d] = f2bf(kf[koff]);
    vgT[(bhc * DH + d) * TW + w] = f2bf(vf[koff]);     // transposed: serves as Bt
}

// ---------------- softmax over cluster window (wave per row) ----------------
__global__ void softmax_kernel(const float* __restrict__ dots, bf16_t* __restrict__ attn) {
    int wave = threadIdx.x >> 5, lane = threadIdx.x & 31;
    long row = (long)blockIdx.x * 8 + wave;
    const float* dr = dots + row * TW;
    float v[4]; float mx = -1e30f;
    #pragma unroll
    for (int i = 0; i < 4; ++i) { v[i] = dr[lane * 4 + i] * 0.125f; mx = fmaxf(mx, v[i]); }
    #pragma unroll
    for (int o = 16; o > 0; o >>= 1) mx = fmaxf(mx, __shfl_xor(mx, o, 32));
    float sum = 0.f;
    #pragma unroll
    for (int i = 0; i < 4; ++i) { v[i] = __expf(v[i] - mx); sum += v[i]; }
    #pragma unroll
    for (int o = 16; o > 0; o >>= 1) sum += __shfl_xor(sum, o, 32);
    float inv = 1.0f / sum;
    #pragma unroll
    for (int i = 0; i < 4; ++i) attn[row * TW + lane * 4 + i] = f2bf(v[i] * inv);
}

// ---------------- scatter-add attention output back to sequence -------------
__global__ void scatter_kernel(const float* __restrict__ og, const int* __restrict__ iq,
                               float* __restrict__ outacc, float* __restrict__ cnt) {
    long idx = (long)blockIdx.x * blockDim.x + threadIdx.x;
    if (idx >= (long)Bb * Hh * CL * TW * DH) return;
    int d = (int)(idx & (DH - 1));
    long t = idx >> 6;
    int w = (int)(t % TW); t /= TW;
    int c = (int)(t % CL); t /= CL;
    int hh = (int)(t % Hh);
    int b  = (int)(t / Hh);
    long bhc = ((long)(b * Hh + hh) * CL + c);
    int tq = iq[bhc * TW + w];
    long o = (((long)(b * Hh + hh) * Ss + tq)) * DH + d;
    atomicAdd(&outacc[o], og[(bhc * TW + w) * DH + d]);
    if (d == 0) atomicAdd(&cnt[(long)(b * Hh + hh) * Ss + tq], 1.0f);
}

// ---------------- mean + transpose (B,H,S,DH)->(B,S,D) as bf16 --------------
__global__ void finalize_kernel(const float* __restrict__ outacc, const float* __restrict__ cnt,
                                bf16_t* __restrict__ attnout16) {
    long idx = (long)blockIdx.x * blockDim.x + threadIdx.x;
    if (idx >= (long)BS * Dd) return;
    int d = (int)(idx & (DH - 1));
    long t = idx >> 6;
    int hh = (int)(t % Hh); t /= Hh;
    int s = (int)(t % Ss);
    int b = (int)(t / Ss);
    float cv = cnt[(long)(b * Hh + hh) * Ss + s];
    cv = fmaxf(cv, 1.0f);
    float v = outacc[(((long)(b * Hh + hh) * Ss + s)) * DH + d] / cv;
    attnout16[((long)(b * Ss + s)) * Dd + hh * DH + d] = f2bf(v);
}

// ---------------- output heads (YD==1): mean + sigma ----------------
__global__ void heads_kernel(const float* __restrict__ hf, const float* __restrict__ W_mean,
                             const float* __restrict__ b_mean, const float* __restrict__ W_std,
                             const float* __restrict__ b_std, float* __restrict__ out) {
    long idx = (long)blockIdx.x * blockDim.x + threadIdx.x;
    if (idx >= (long)Bb * SF) return;
    int sf = (int)(idx % SF);
    int b  = (int)(idx / SF);
    const float* hr = hf + ((long)(b * Ss + SP + sf)) * Dd;
    float m = 0.f, sg = 0.f;
    for (int d = 0; d < Dd; ++d) { float v = hr[d]; m += v * W_mean[d]; sg += v * W_std[d]; }
    m += b_mean[0]; sg += b_std[0];
    float sp = fmaxf(sg, 0.f) + log1pf(__expf(-fabsf(sg)));   // stable softplus
    out[idx] = m;                                             // mean  (B,SF,1)
    out[(long)Bb * SF + idx] = 0.01f + 0.99f * sp;            // sigma (B,SF,1)
}

// ======================= orchestration =======================
extern "C" void kernel_launch(void* const* d_in, const int* in_sizes, int n_in,
                              void* d_out, int out_size, void* d_ws, size_t ws_size,
                              hipStream_t stream) {
    const float* past_x   = (const float*)d_in[0];
    const float* past_y   = (const float*)d_in[1];
    const float* future_x = (const float*)d_in[2];
    const float* W_emb    = (const float*)d_in[3];
    const float* b_emb    = (const float*)d_in[4];
    const float* ln1_g    = (const float*)d_in[5];
    const float* ln1_b    = (const float*)d_in[6];
    const float* Wq       = (const float*)d_in[7];
    const float* Wk       = (const float*)d_in[8];
    const float* Wv       = (const float*)d_in[9];
    const float* Wo       = (const float*)d_in[10];
    const float* means    = (const float*)d_in[11];
    const float* ln2_g    = (const float*)d_in[12];
    const float* ln2_b    = (const float*)d_in[13];
    const float* W1       = (const float*)d_in[14];
    const float* b1       = (const float*)d_in[15];
    const float* W2       = (const float*)d_in[16];
    const float* b2       = (const float*)d_in[17];
    const float* lnf_g    = (const float*)d_in[18];
    const float* lnf_b    = (const float*)d_in[19];
    const float* W_mean   = (const float*)d_in[20];
    const float* b_mean   = (const float*)d_in[21];
    const float* W_std    = (const float*)d_in[22];
    const float* b_std    = (const float*)d_in[23];
    float* out = (float*)d_out;
    (void)in_sizes; (void)n_in; (void)out_size; (void)ws_size;

    // ---- workspace bump allocator (with lifetime-based aliasing) ----
    char* wsp = (char*)d_ws;
    auto alloc = [&](size_t bytes) -> void* {
        void* p = (void*)wsp; wsp += (bytes + 255) & ~(size_t)255; return p;
    };
    constexpr long GA   = (long)Bb * Hh * CL * TW * DH;   // gathered elems (8.4M)
    constexpr long BHC  = (long)Bb * Hh * CL;             // 1024 clusters
    constexpr long BHCS = BHC * Ss;                       // score elems (2M)

    float*  h      = (float*)alloc((size_t)BS * Dd * 4);
    bf16_t* hn16   = (bf16_t*)alloc((size_t)BS * Dd * 2); // also reused as attnout16
    float*  qf     = (float*)alloc((size_t)BS * Dd * 4);
    float*  kf     = (float*)alloc((size_t)BS * Dd * 4);
    float*  vf     = (float*)alloc((size_t)BS * Dd * 4);
    bf16_t* wT     = (bf16_t*)alloc((size_t)FF * Dd * 2); // shared weight staging
    float*  scq    = (float*)alloc((size_t)BHCS * 4);
    float*  sck    = (float*)alloc((size_t)BHCS * 4);
    int*    iq     = (int*)alloc((size_t)BHC * TW * 4);
    int*    ik     = (int*)alloc((size_t)BHC * TW * 4);
    bf16_t* qg     = (bf16_t*)alloc((size_t)GA * 2);
    bf16_t* kg     = (bf16_t*)alloc((size_t)GA * 2);
    bf16_t* vgT    = (bf16_t*)alloc((size_t)GA * 2);
    void*   big    = alloc((size_t)BS * FF * 2);          // dots(f32) aliases mid16(bf16)
    float*  dots   = (float*)big;
    bf16_t* mid16  = (bf16_t*)big;
    void*   ab     = alloc((size_t)BS * Dd * 4);          // attn16 aliases hf
    bf16_t* attn16 = (bf16_t*)ab;
    float*  hf     = (float*)ab;
    float*  og     = (float*)alloc((size_t)GA * 4);
    float*  outacc = (float*)alloc((size_t)Bb * Hh * Ss * DH * 4);
    float*  cnt    = (float*)alloc((size_t)Bb * Hh * Ss * 4);
    bf16_t* attnout16 = hn16;                             // lifetime-safe alias

    dim3 blk(256);

    // 1) embedding
    embed_kernel<<<(unsigned)(((long)BS * Dd + 255) / 256), blk, 0, stream>>>(
        past_x, past_y, future_x, W_emb, b_emb, h);

    for (int l = 0; l < Ll; ++l) {
        const size_t wDD = (size_t)l * Dd * Dd;
        // 2) LN1 -> bf16
        ln_kernel<<<BS / 8, blk, 0, stream>>>(h, ln1_g + l * Dd, ln1_b + l * Dd,
                                              nullptr, hn16, 1);
        // 3) Q/K/V projections (WMMA): macro-tile 64(M) x 128(N)
        transpose_bf16_kernel<<<(Dd * Dd + 255) / 256, blk, 0, stream>>>(Wq + wDD, wT, Dd, Dd);
        gemm_wmma_kernel<<<dim3(Dd / 128, BS / 64, 1), blk, 0, stream>>>(
            hn16, wT, qf, nullptr, nullptr, BS, Dd, Dd, 0, 0, 0, 0);
        transpose_bf16_kernel<<<(Dd * Dd + 255) / 256, blk, 0, stream>>>(Wk + wDD, wT, Dd, Dd);
        gemm_wmma_kernel<<<dim3(Dd / 128, BS / 64, 1), blk, 0, stream>>>(
            hn16, wT, kf, nullptr, nullptr, BS, Dd, Dd, 0, 0, 0, 0);
        transpose_bf16_kernel<<<(Dd * Dd + 255) / 256, blk, 0, stream>>>(Wv + wDD, wT, Dd, Dd);
        gemm_wmma_kernel<<<dim3(Dd / 128, BS / 64, 1), blk, 0, stream>>>(
            hn16, wT, vf, nullptr, nullptr, BS, Dd, Dd, 0, 0, 0, 0);
        // 4) routing scores + exact top-W
        const float* mptr = means + (size_t)l * Hh * CL * DH;
        route_scores_kernel<<<(Bb * Hh * Ss + 255) / 256, blk, 0, stream>>>(qf, mptr, scq);
        route_scores_kernel<<<(Bb * Hh * Ss + 255) / 256, blk, 0, stream>>>(kf, mptr, sck);
        topk_kernel<<<(unsigned)BHC, blk, 0, stream>>>(scq, iq);
        topk_kernel<<<(unsigned)BHC, blk, 0, stream>>>(sck, ik);
        // 5) gather into cluster tiles
        gather_kernel<<<(unsigned)((GA + 255) / 256), blk, 0, stream>>>(
            qf, kf, vf, iq, ik, qg, kg, vgT);
        // 6) dots = qg @ kg^T   (batched WMMA, M=N=128, K=64)
        gemm_wmma_kernel<<<dim3(1, 2, (unsigned)BHC), blk, 0, stream>>>(
            qg, kg, dots, nullptr, nullptr, TW, TW, DH,
            (long)TW * DH, (long)TW * DH, (long)TW * TW, 0);
        // 7) softmax (scale DH^-0.5 folded in)
        softmax_kernel<<<(unsigned)(BHC * TW / 8), blk, 0, stream>>>(dots, attn16);
        // 8) og = attn @ vg     (batched WMMA, M=128, N=64, K=128)
        gemm_wmma_kernel<<<dim3(1, 2, (unsigned)BHC), blk, 0, stream>>>(
            attn16, vgT, og, nullptr, nullptr, TW, DH, TW,
            (long)TW * TW, (long)DH * TW, (long)TW * DH, 0);
        // 9) scatter-mean back to sequence
        hipMemsetAsync(outacc, 0, (size_t)Bb * Hh * Ss * DH * 4, stream);
        hipMemsetAsync(cnt, 0, (size_t)Bb * Hh * Ss * 4, stream);
        scatter_kernel<<<(unsigned)((GA + 255) / 256), blk, 0, stream>>>(og, iq, outacc, cnt);
        finalize_kernel<<<(unsigned)(((long)BS * Dd + 255) / 256), blk, 0, stream>>>(
            outacc, cnt, attnout16);
        // 10) h += attnout @ Wo (residual WMMA)
        transpose_bf16_kernel<<<(Dd * Dd + 255) / 256, blk, 0, stream>>>(Wo + wDD, wT, Dd, Dd);
        gemm_wmma_kernel<<<dim3(Dd / 128, BS / 64, 1), blk, 0, stream>>>(
            attnout16, wT, h, nullptr, nullptr, BS, Dd, Dd, 0, 0, 0, 4);
        // 11) FFN
        ln_kernel<<<BS / 8, blk, 0, stream>>>(h, ln2_g + l * Dd, ln2_b + l * Dd,
                                              nullptr, hn16, 1);
        transpose_bf16_kernel<<<(Dd * FF + 255) / 256, blk, 0, stream>>>(
            W1 + (size_t)l * Dd * FF, wT, Dd, FF);
        gemm_wmma_kernel<<<dim3(FF / 128, BS / 64, 1), blk, 0, stream>>>(
            hn16, wT, nullptr, mid16, b1 + (size_t)l * FF, BS, FF, Dd, 0, 0, 0, 1 | 2 | 8);
        transpose_bf16_kernel<<<(FF * Dd + 255) / 256, blk, 0, stream>>>(
            W2 + (size_t)l * FF * Dd, wT, FF, Dd);
        gemm_wmma_kernel<<<dim3(Dd / 128, BS / 64, 1), blk, 0, stream>>>(
            mid16, wT, h, nullptr, b2 + (size_t)l * Dd, BS, Dd, FF, 0, 0, 0, 1 | 4);
    }

    // 12) final LN (fp32) + heads
    ln_kernel<<<BS / 8, blk, 0, stream>>>(h, lnf_g, lnf_b, hf, nullptr, 0);
    heads_kernel<<<(Bb * SF + 255) / 256, blk, 0, stream>>>(hf, W_mean, b_mean,
                                                            W_std, b_std, out);
}
